// TransformerXL_12008728559652
// MI455X (gfx1250) — matrix operand order repaired and checked by reading, not compiled, not run on hardware
//
#include <hip/hip_runtime.h>
#include <hip/hip_bf16.h>

// Transformer-XL forward for MI455X (gfx1250, wave32, WMMA).
// All GEMMs (QKV, r-proj, W_o, FF1, FF2) and attention (AC+BD scores, P@V)
// run on v_wmma_f32_16x16x32_bf16 with fp32 accumulation. QKV/KR are produced
// in bf16 so attention K/KR tiles stream via GLOBAL_LOAD_ASYNC_TO_LDS (ASYNCcnt)
// when the toolchain exposes the builtin.

#define L_  4
#define H_  16
#define D_  1024
#define DH_ 64
#define DF_ 4096
#define B_  4
#define S_  1024

typedef __attribute__((ext_vector_type(16))) __bf16 v16bf;
typedef __attribute__((ext_vector_type(8)))  float  v8f;
typedef __attribute__((ext_vector_type(4)))  int    v4i;

#define AS_G __attribute__((address_space(1)))
#define AS_L __attribute__((address_space(3)))

#if defined(__has_builtin)
#  if __has_builtin(__builtin_amdgcn_global_load_async_to_lds_b128)
#    define HAVE_ASYNC_LDS 1
#  endif
#  if __has_builtin(__builtin_amdgcn_s_wait_asynccnt)
#    define ASYNC_WAIT() __builtin_amdgcn_s_wait_asynccnt(0)
#  endif
#endif
#ifndef ASYNC_WAIT
#  define ASYNC_WAIT() asm volatile("s_wait_asynccnt 0" ::: "memory")
#endif

union FragAB { v16bf v; unsigned int u[8]; };

__device__ __forceinline__ unsigned int bf16_1(float a) {
  unsigned int x = __float_as_uint(a);
  return (x + 0x7FFFu + ((x >> 16) & 1u)) >> 16;   // round-to-nearest-even
}
__device__ __forceinline__ unsigned int bf16_2(float a, float b) {
  return bf16_1(a) | (bf16_1(b) << 16);
}
__device__ __forceinline__ float bf16_lo_f(unsigned int p) { return __uint_as_float(p << 16); }
__device__ __forceinline__ float bf16_hi_f(unsigned int p) { return __uint_as_float(p & 0xFFFF0000u); }

#define WMMA_BF16(a, b, c) \
  __builtin_amdgcn_wmma_f32_16x16x32_bf16(false, (a), false, (b), (short)0, (c), false, false)

// ---- WMMA fragment loaders (LDS stored contraction-contiguous, padded stride) ----
__device__ __forceinline__ v16bf frag_a_lds(const unsigned short* lds, int row0, int stride,
                                            int k0, int lane) {
  const int m = lane & 15, hi = (lane >> 4) & 1;
  const unsigned int* p32 = (const unsigned int*)lds;
  const int base = (row0 + m) * stride;
  FragAB f;
#pragma unroll
  for (int i = 0; i < 4; ++i) {
    f.u[i]     = p32[(base + k0 + hi * 8 + 2 * i) >> 1];
    f.u[i + 4] = p32[(base + k0 + 16 + hi * 8 + 2 * i) >> 1];
  }
  return f.v;
}
__device__ __forceinline__ v16bf frag_b_lds(const unsigned short* lds, int col0, int stride,
                                            int k0, int lane) {
  const int n = col0 + (lane & 15), hi = (lane >> 4) & 1;
  const unsigned int* p32 = (const unsigned int*)lds;
  const int base = n * stride;
  FragAB f;
#pragma unroll
  for (int i = 0; i < 8; ++i)
    f.u[i] = p32[(base + k0 + ((i < 4) ? (hi * 8 + 2 * i) : (16 + hi * 8 + 2 * (i - 4)))) >> 1];
  return f.v;
}
__device__ __forceinline__ v16bf frag_a_f32_add(const float* lds, int row0, int stride, int k0,
                                                const float* __restrict__ add, int lane) {
  const int m = lane & 15, hi = (lane >> 4) & 1;
  const float* rp = lds + (row0 + m) * stride;
  FragAB f;
#pragma unroll
  for (int i = 0; i < 8; ++i) {
    const int k = k0 + ((i < 4) ? (hi * 8 + 2 * i) : (16 + hi * 8 + 2 * (i - 4)));
    f.u[i] = bf16_2(rp[k] + add[k], rp[k + 1] + add[k + 1]);
  }
  return f.v;
}

// =======================  GEMM: C = act(A(MxK) * B(KxN) + bias)  =======================
// Block tile 128x64, 8 waves as 4(M)x2(N), each 32x32. Double-buffered LDS, K-step 64.
__device__ __forceinline__ void gemm_stage(const float* __restrict__ A,
                                           const float* __restrict__ Bm,
                                           unsigned short* lA, unsigned short* lB,
                                           int m0, int n0, int k0, int lda, int ldb, int tid) {
#pragma unroll
  for (int it = 0; it < 8; ++it) {
    int e = (it * 256 + tid) * 4;
    int r = e >> 6, c = e & 63;
    const float4 f = *(const float4*)(A + (long)(m0 + r) * lda + k0 + c);
    unsigned int* dst = (unsigned int*)&lA[r * 68 + c];
    dst[0] = bf16_2(f.x, f.y);
    dst[1] = bf16_2(f.z, f.w);
  }
#pragma unroll
  for (int it = 0; it < 4; ++it) {
    int e = (it * 256 + tid) * 4;
    int r = e >> 6, c = e & 63;
    const float4 f = *(const float4*)(Bm + (long)(k0 + r) * ldb + n0 + c);
    lB[(c + 0) * 68 + r] = (unsigned short)bf16_1(f.x);
    lB[(c + 1) * 68 + r] = (unsigned short)bf16_1(f.y);
    lB[(c + 2) * 68 + r] = (unsigned short)bf16_1(f.z);
    lB[(c + 3) * 68 + r] = (unsigned short)bf16_1(f.w);
  }
}

__global__ __launch_bounds__(256) void gemm_bf16(const float* __restrict__ A,
                                                 const float* __restrict__ Bm,
                                                 const float* __restrict__ bias,
                                                 void* __restrict__ Cout,
                                                 int M, int N, int K, int relu, int out_bf16) {
  __shared__ __align__(16) unsigned short lA[2][128 * 68];
  __shared__ __align__(16) unsigned short lB[2][64 * 68];

  const int tid  = threadIdx.x;
  const int lane = tid & 31;
  const int w    = tid >> 5;
  const int wm   = (w & 3) * 32;
  const int wn   = (w >> 2) * 32;
  const int m0   = blockIdx.x * 128;
  const int n0   = blockIdx.y * 64;

  v8f acc[2][2];
#pragma unroll
  for (int i = 0; i < 2; ++i)
#pragma unroll
    for (int j = 0; j < 2; ++j)
#pragma unroll
      for (int e = 0; e < 8; ++e) acc[i][j][e] = 0.0f;

  const int nk = K >> 6;
  gemm_stage(A, Bm, lA[0], lB[0], m0, n0, 0, K, N, tid);
  __syncthreads();

  for (int kt = 0; kt < nk; ++kt) {
    const int cur = kt & 1;
    if (kt + 1 < nk)
      gemm_stage(A, Bm, lA[cur ^ 1], lB[cur ^ 1], m0, n0, (kt + 1) * 64, K, N, tid);
    if (kt + 2 < nk) {   // global_prefetch_b8 for tile k+2
      __builtin_prefetch(A + (long)(m0 + (tid >> 1)) * K + (kt + 2) * 64 + (tid & 1) * 32, 0, 0);
      __builtin_prefetch(Bm + (long)((kt + 2) * 64 + (tid >> 2)) * N + n0 + (tid & 3) * 16, 0, 0);
    }
#pragma unroll
    for (int ks = 0; ks < 64; ks += 32) {
      v16bf a0 = frag_a_lds(lA[cur], wm,      68, ks, lane);
      v16bf a1 = frag_a_lds(lA[cur], wm + 16, 68, ks, lane);
      v16bf b0 = frag_b_lds(lB[cur], wn,      68, ks, lane);
      v16bf b1 = frag_b_lds(lB[cur], wn + 16, 68, ks, lane);
      acc[0][0] = WMMA_BF16(a0, b0, acc[0][0]);
      acc[0][1] = WMMA_BF16(a0, b1, acc[0][1]);
      acc[1][0] = WMMA_BF16(a1, b0, acc[1][0]);
      acc[1][1] = WMMA_BF16(a1, b1, acc[1][1]);
    }
    __syncthreads();
  }

  const int hi = (lane >> 4) & 1, ln15 = lane & 15;
  if (out_bf16) {
    unsigned short* C = (unsigned short*)Cout;
#pragma unroll
    for (int ti = 0; ti < 2; ++ti)
#pragma unroll
      for (int tj = 0; tj < 2; ++tj) {
        const int col = n0 + wn + tj * 16 + ln15;
        const float bv = bias ? bias[col] : 0.0f;
#pragma unroll
        for (int vv = 0; vv < 8; ++vv) {
          const int row = m0 + wm + ti * 16 + vv + hi * 8;
          float val = acc[ti][tj][vv] + bv;
          if (relu) val = fmaxf(val, 0.0f);
          C[(long)row * N + col] = (unsigned short)bf16_1(val);
        }
      }
  } else {
    float* C = (float*)Cout;
#pragma unroll
    for (int ti = 0; ti < 2; ++ti)
#pragma unroll
      for (int tj = 0; tj < 2; ++tj) {
        const int col = n0 + wn + tj * 16 + ln15;
        const float bv = bias ? bias[col] : 0.0f;
#pragma unroll
        for (int vv = 0; vv < 8; ++vv) {
          const int row = m0 + wm + ti * 16 + vv + hi * 8;
          float val = acc[ti][tj][vv] + bv;
          if (relu) val = fmaxf(val, 0.0f);
          C[(long)row * N + col] = val;
        }
      }
  }
}

// =======================  Relative attention (flash-style, bf16 inputs)  =======================
// Grid: (S/128, B*H). 8 waves, 16 q-rows each; sweeps 32 k-positions per tile.
// K/KR tiles stream via async global->LDS DMA when available.
__global__ __launch_bounds__(256) void attn_kernel(const unsigned short* __restrict__ qkv16,
                                                   const unsigned short* __restrict__ kr16,
                                                   const float* __restrict__ u,
                                                   const float* __restrict__ v,
                                                   float* __restrict__ vec) {
  __shared__ __align__(16) float          lQ[128 * 66];    // q tile fp32, stride 66
  __shared__ __align__(16) unsigned short lK[32 * 72];     // k tile: pos x dh, stride 72 (16B rows)
  __shared__ __align__(16) unsigned short lR[32 * 72];     // kr tile
  __shared__ __align__(16) unsigned short lV[64 * 36];     // v tile transposed: dh x pos
  __shared__ __align__(16) unsigned short lP[8 * 16 * 36]; // per-wave P (16 x 32)

  const int tid  = threadIdx.x;
  const int lane = tid & 31;
  const int w    = tid >> 5;
  const int hi   = (lane >> 4) & 1, ln15 = lane & 15;
  const int q0   = blockIdx.x * 128;
  const int head = blockIdx.y & (H_ - 1);
  const int bb   = blockIdx.y >> 4;

  const unsigned short* qb16 = qkv16 + (long)bb * S_ * (3 * H_ * DH_) + head * DH_;
  const unsigned short* kb16 = qb16 + H_ * DH_;
  const unsigned short* vb16 = qb16 + 2 * H_ * DH_;
  const unsigned short* rb16 = kr16 + head * DH_;
  const float* uvec = u + head * DH_;
  const float* vvec = v + head * DH_;

  // Stage the 128x64 q tile (bf16 -> fp32).
#pragma unroll
  for (int it = 0; it < 8; ++it) {
    int e = (it * 256 + tid) * 4;
    int r = e >> 6, c = e & 63;
    const uint2 g = *(const uint2*)(qb16 + (long)(q0 + r) * (3 * H_ * DH_) + c);
    float* dst = &lQ[r * 66 + c];
    dst[0] = bf16_lo_f(g.x); dst[1] = bf16_hi_f(g.x);
    dst[2] = bf16_lo_f(g.y); dst[3] = bf16_hi_f(g.y);
  }
  __syncthreads();

  const int wrow = w * 16;
  v16bf qu0 = frag_a_f32_add(lQ, wrow, 66, 0,  uvec, lane);
  v16bf qu1 = frag_a_f32_add(lQ, wrow, 66, 32, uvec, lane);
  v16bf qv0 = frag_a_f32_add(lQ, wrow, 66, 0,  vvec, lane);
  v16bf qv1 = frag_a_f32_add(lQ, wrow, 66, 32, vvec, lane);

  float mrow[8], lrow[8];
  v8f o[4];
#pragma unroll
  for (int i = 0; i < 8; ++i) { mrow[i] = -3.0e38f; lrow[i] = 0.0f; }
#pragma unroll
  for (int n = 0; n < 4; ++n)
#pragma unroll
    for (int e = 0; e < 8; ++e) o[n][e] = 0.0f;

  for (int kt = 0; kt < S_; kt += 32) {
    __syncthreads();   // previous tile fully consumed
    // K/KR tiles: each thread moves one aligned 16B chunk (8 bf16).
    {
      const int p = tid >> 3, c8 = (tid & 7) * 8;
#if defined(HAVE_ASYNC_LDS)
      AS_G v4i* gk = (AS_G v4i*)(kb16 + (long)(kt + p) * (3 * H_ * DH_) + c8);
      AS_G v4i* gr = (AS_G v4i*)(rb16 + (long)(kt + p) * (H_ * DH_) + c8);
      AS_L v4i* dk = (AS_L v4i*)&lK[p * 72 + c8];
      AS_L v4i* dr = (AS_L v4i*)&lR[p * 72 + c8];
      __builtin_amdgcn_global_load_async_to_lds_b128(gk, dk, 0, 0);
      __builtin_amdgcn_global_load_async_to_lds_b128(gr, dr, 0, 0);
#else
      *(uint4*)&lK[p * 72 + c8] = *(const uint4*)(kb16 + (long)(kt + p) * (3 * H_ * DH_) + c8);
      *(uint4*)&lR[p * 72 + c8] = *(const uint4*)(rb16 + (long)(kt + p) * (H_ * DH_) + c8);
#endif
    }
    // V tile transposed (dh x pos).
#pragma unroll
    for (int it = 0; it < 2; ++it) {
      int e = (it * 256 + tid) * 4;
      int p = e >> 6, c = e & 63;
      const uint2 g = *(const uint2*)(vb16 + (long)(kt + p) * (3 * H_ * DH_) + c);
      lV[(c + 0) * 36 + p] = (unsigned short)(g.x & 0xFFFFu);
      lV[(c + 1) * 36 + p] = (unsigned short)(g.x >> 16);
      lV[(c + 2) * 36 + p] = (unsigned short)(g.y & 0xFFFFu);
      lV[(c + 3) * 36 + p] = (unsigned short)(g.y >> 16);
    }
#if defined(HAVE_ASYNC_LDS)
    ASYNC_WAIT();
#endif
    __syncthreads();

    // Scores: AC + BD, scaled by 1/sqrt(DH)=0.125.
    v8f sc[2];
#pragma unroll
    for (int nh = 0; nh < 2; ++nh) {
      v8f a;
#pragma unroll
      for (int e = 0; e < 8; ++e) a[e] = 0.0f;
      a = WMMA_BF16(qu0, frag_b_lds(lK, nh * 16, 72, 0,  lane), a);
      a = WMMA_BF16(qu1, frag_b_lds(lK, nh * 16, 72, 32, lane), a);
      a = WMMA_BF16(qv0, frag_b_lds(lR, nh * 16, 72, 0,  lane), a);
      a = WMMA_BF16(qv1, frag_b_lds(lR, nh * 16, 72, 32, lane), a);
#pragma unroll
      for (int e = 0; e < 8; ++e) sc[nh][e] = a[e] * 0.125f;
    }

    // Online softmax; row M = vv + 8*hi lives in one half-wave -> width-16 reductions.
    unsigned short* Pw = &lP[w * 16 * 36];
#pragma unroll
    for (int vv = 0; vv < 8; ++vv) {
      float tm = fmaxf(sc[0][vv], sc[1][vv]);
      tm = fmaxf(tm, __shfl_xor(tm, 1, 16));
      tm = fmaxf(tm, __shfl_xor(tm, 2, 16));
      tm = fmaxf(tm, __shfl_xor(tm, 4, 16));
      tm = fmaxf(tm, __shfl_xor(tm, 8, 16));
      const float mn    = fmaxf(mrow[vv], tm);
      const float scale = __expf(mrow[vv] - mn);
      const float p0 = __expf(sc[0][vv] - mn);
      const float p1 = __expf(sc[1][vv] - mn);
      float rs = p0 + p1;
      rs += __shfl_xor(rs, 1, 16);
      rs += __shfl_xor(rs, 2, 16);
      rs += __shfl_xor(rs, 4, 16);
      rs += __shfl_xor(rs, 8, 16);
      lrow[vv] = lrow[vv] * scale + rs;
      mrow[vv] = mn;
      o[0][vv] *= scale; o[1][vv] *= scale; o[2][vv] *= scale; o[3][vv] *= scale;
      const int prow = vv + hi * 8;
      Pw[prow * 36 + ln15]      = (unsigned short)bf16_1(p0);
      Pw[prow * 36 + 16 + ln15] = (unsigned short)bf16_1(p1);
    }
    // Same-wave LDS RAW across lanes: LDS in-order per wave; fence the compiler too.
    asm volatile("s_wait_dscnt 0" ::: "memory");

    v16bf pf = frag_a_lds(Pw, 0, 36, 0, lane);
#pragma unroll
    for (int n = 0; n < 4; ++n) {
      v16bf vb = frag_b_lds(lV, n * 16, 36, 0, lane);
      o[n] = WMMA_BF16(pf, vb, o[n]);
    }
  }

  // Normalize and write vec[b, q, h*64 + d] (fp32).
#pragma unroll
  for (int vv = 0; vv < 8; ++vv) {
    const float inv = 1.0f / lrow[vv];
    const int row = q0 + wrow + vv + hi * 8;
#pragma unroll
    for (int n = 0; n < 4; ++n)
      vec[((long)bb * S_ + row) * (H_ * DH_) + head * DH_ + n * 16 + ln15] = o[n][vv] * inv;
  }
}

// =======================  LayerNorm (+optional residual)  =======================
__global__ __launch_bounds__(256) void ln_kernel(const float* __restrict__ in,
                                                 const float* __restrict__ res,
                                                 const float* __restrict__ g,
                                                 const float* __restrict__ b,
                                                 float* __restrict__ out) {
  __shared__ float red[8];
  const int row = blockIdx.x;
  const int tid = threadIdx.x;
  const float* ip = in + (long)row * D_;
  const float* rp = res ? res + (long)row * D_ : nullptr;
  float x[4];
  float s = 0.0f;
#pragma unroll
  for (int i = 0; i < 4; ++i) {
    const int c = tid + i * 256;
    x[i] = ip[c] + (rp ? rp[c] : 0.0f);
    s += x[i];
  }
#pragma unroll
  for (int off = 16; off >= 1; off >>= 1) s += __shfl_xor(s, off, 32);
  if ((tid & 31) == 0) red[tid >> 5] = s;
  __syncthreads();
  float tot = 0.0f;
#pragma unroll
  for (int i = 0; i < 8; ++i) tot += red[i];
  const float mu = tot * (1.0f / D_);
  __syncthreads();
  float vs = 0.0f;
#pragma unroll
  for (int i = 0; i < 4; ++i) { const float d = x[i] - mu; vs += d * d; }
#pragma unroll
  for (int off = 16; off >= 1; off >>= 1) vs += __shfl_xor(vs, off, 32);
  if ((tid & 31) == 0) red[tid >> 5] = vs;
  __syncthreads();
  float vt = 0.0f;
#pragma unroll
  for (int i = 0; i < 8; ++i) vt += red[i];
  const float inv = rsqrtf(vt * (1.0f / D_) + 1e-5f);
#pragma unroll
  for (int i = 0; i < 4; ++i) {
    const int c = tid + i * 256;
    out[(long)row * D_ + c] = (x[i] - mu) * inv * g[c] + b[c];
  }
}

// =======================  Positional encoding  =======================
__global__ __launch_bounds__(256) void posenc_kernel(float* __restrict__ r) {
  const int row = blockIdx.x;
  const float pos = (float)(S_ - 1 - row);
#pragma unroll
  for (int j = 0; j < 2; ++j) {
    const int i = threadIdx.x + j * 256;                 // i in [0, D/2)
    const float freq = __expf(-((float)(2 * i) / (float)D_) * 9.210340371976184f); // ln(1e4)
    const float ang = pos * freq;
    r[(long)row * D_ + i]          = __sinf(ang);
    r[(long)row * D_ + D_ / 2 + i] = __cosf(ang);
  }
}

// =======================  Host orchestration  =======================
extern "C" void kernel_launch(void* const* d_in, const int* in_sizes, int n_in,
                              void* d_out, int out_size, void* d_ws, size_t ws_size,
                              hipStream_t stream) {
  const float* x     = (const float*)d_in[0];
  const float* W_qkv = (const float*)d_in[1];
  const float* W_r   = (const float*)d_in[2];
  const float* W_o   = (const float*)d_in[3];
  const float* u     = (const float*)d_in[4];
  const float* v     = (const float*)d_in[5];
  const float* ln1g  = (const float*)d_in[6];
  const float* ln1b  = (const float*)d_in[7];
  const float* Wff1  = (const float*)d_in[8];
  const float* bff1  = (const float*)d_in[9];
  const float* Wff2  = (const float*)d_in[10];
  const float* bff2  = (const float*)d_in[11];
  const float* ln2g  = (const float*)d_in[12];
  const float* ln2b  = (const float*)d_in[13];

  float* ws = (float*)d_ws;
  size_t off = 0;
  float* h    = ws + off; off += (size_t)B_ * S_ * D_;          // 16 MB
  float* r    = ws + off; off += (size_t)S_ * D_;               // 4 MB
  float* vec  = ws + off; off += (size_t)B_ * S_ * D_;          // 16 MB
  float* tmp  = ws + off; off += (size_t)B_ * S_ * D_;          // 16 MB
  float* ff1  = ws + off; off += (size_t)B_ * S_ * DF_;         // 64 MB
  unsigned short* qkv16 = (unsigned short*)(ws + off);          // 24 MB (bf16)
  off += (size_t)B_ * S_ * 3 * D_ / 2;
  unsigned short* kr16  = (unsigned short*)(ws + off);          // 2 MB (bf16)

  (void)hipMemcpyAsync(h, x, (size_t)B_ * S_ * D_ * sizeof(float),
                       hipMemcpyDeviceToDevice, stream);
  posenc_kernel<<<S_, 256, 0, stream>>>(r);

  const int MROWS = B_ * S_;   // 4096
  for (int i = 0; i < L_; ++i) {
    // qkv16 = bf16(h @ W_qkv[i])   (4096 x 1024 x 3072)
    gemm_bf16<<<dim3(MROWS / 128, (3 * D_) / 64), 256, 0, stream>>>(
        h, W_qkv + (size_t)i * D_ * 3 * D_, nullptr, qkv16, MROWS, 3 * D_, D_, 0, 1);
    // kr16 = bf16(r @ W_r[i])      (1024 x 1024 x 1024)
    gemm_bf16<<<dim3(S_ / 128, D_ / 64), 256, 0, stream>>>(
        r, W_r + (size_t)i * D_ * D_, nullptr, kr16, S_, D_, D_, 0, 1);
    // attention -> vec (fp32)
    attn_kernel<<<dim3(S_ / 128, B_ * H_), 256, 0, stream>>>(qkv16, kr16, u, v, vec);
    // tmp = vec @ W_o[i]
    gemm_bf16<<<dim3(MROWS / 128, D_ / 64), 256, 0, stream>>>(
        vec, W_o + (size_t)i * D_ * D_, nullptr, tmp, MROWS, D_, D_, 0, 0);
    // h = LN(h + tmp)
    ln_kernel<<<MROWS, 256, 0, stream>>>(tmp, h, ln1g + (size_t)i * D_,
                                         ln1b + (size_t)i * D_, h);
    // ff1 = relu(h @ W_ff1[i] + b_ff1[i])   (4096 x 1024 x 4096)
    gemm_bf16<<<dim3(MROWS / 128, DF_ / 64), 256, 0, stream>>>(
        h, Wff1 + (size_t)i * D_ * DF_, bff1 + (size_t)i * DF_, ff1, MROWS, DF_, D_, 1, 0);
    // tmp = ff1 @ W_ff2[i] + b_ff2[i]       (4096 x 4096 x 1024)
    gemm_bf16<<<dim3(MROWS / 128, D_ / 64), 256, 0, stream>>>(
        ff1, Wff2 + (size_t)i * DF_ * D_, bff2 + (size_t)i * D_, tmp, MROWS, D_, DF_, 0, 0);
    // h = LN(tmp)  (no residual, per reference)
    ln_kernel<<<MROWS, 256, 0, stream>>>(tmp, nullptr, ln2g + (size_t)i * D_,
                                         ln2b + (size_t)i * D_, h);
  }
  (void)hipMemcpyAsync(d_out, h, (size_t)B_ * S_ * D_ * sizeof(float),
                       hipMemcpyDeviceToDevice, stream);
}
